// RelativeMultiHeadAttn_16226386444355
// MI455X (gfx1250) — compile-verified, matrix operand order checked by reading, not run on hardware
//
#include <hip/hip_runtime.h>
#include <math.h>

typedef __attribute__((ext_vector_type(16))) _Float16 v16h;
typedef __attribute__((ext_vector_type(8)))  float    v8f;

#if __has_builtin(__builtin_amdgcn_global_load_async_to_lds_b128) && \
    __has_builtin(__builtin_amdgcn_s_wait_asynccnt)
#define ASYNC_LDS 1
#endif

#ifdef ASYNC_LDS
typedef int v4i_ __attribute__((vector_size(16)));
typedef __attribute__((address_space(1))) v4i_ GV4;   // global int4
typedef __attribute__((address_space(3))) v4i_ LV4;   // LDS int4
#endif

// 16B global -> LDS copy; async (ASYNCcnt) via GLOBAL_LOAD_ASYNC_TO_LDS_B128.
__device__ __forceinline__ void cp16(void* lds, const void* g) {
#ifdef ASYNC_LDS
  __builtin_amdgcn_global_load_async_to_lds_b128((GV4*)g, (LV4*)lds, 0, 0);
#else
  *(uint4*)lds = *(const uint4*)g;
#endif
}

__device__ __forceinline__ void cp_fence() {
#ifdef ASYNC_LDS
  __builtin_amdgcn_s_wait_asynccnt(0);
#endif
}

__device__ __forceinline__ v8f wmma16(v16h a, v16h b, v8f c) {
  // v_wmma_f32_16x16x32_f16: (neg_a, A, neg_b, B, c_mod, C, reuse_a, reuse_b)
  return __builtin_amdgcn_wmma_f32_16x16x32_f16(false, a, false, b, (short)0, c, false, false);
}

// A/B fragment loader for v_wmma_f32_16x16x32_f16 from row-major LDS.
// 'row' = matrix row (A) or output column (B). K striping per ISA 7.12.2:
// lane<16 holds K {0..7,16..23}; lane>=16 holds K {8..15,24..31}; contiguous pairs per VGPR.
__device__ __forceinline__ v16h load_frag(const _Float16* base, int row, int stride, int lane) {
  v16h f;
  const _Float16* p = base + row * stride;
  const int lhi = (lane >> 4) << 3;
#pragma unroll
  for (int v = 0; v < 8; ++v) {
    const int kb = ((v & 4) << 2) + lhi + ((v & 3) << 1);
    f[2 * v]     = p[kb];
    f[2 * v + 1] = p[kb + 1];
  }
  return f;
}

// ---- positional embedding: posh[l][d], l in [0,1024), relative position = l-512 ----
__global__ void pos_kernel(_Float16* __restrict__ posh) {
  const int l = blockIdx.x;   // 0..1023
  const int d = threadIdx.x;  // 0..63
  const float p = (float)(l - 512);
  const int dd = d & 31;
  const float inv = expf(-(float)dd * (logf(10000.f) / 31.f));
  const float ang = p * inv;
  const float v = (d < 32) ? sinf(ang) : cosf(ang);
  posh[l * 64 + d] = (_Float16)v;
}

__global__ void cvt_kernel(const float* __restrict__ x, const float* __restrict__ w,
                           _Float16* __restrict__ xh, _Float16* __restrict__ wh) {
  const int i = blockIdx.x * 256 + threadIdx.x;
  if (i < 8 * 512 * 512) xh[i] = (_Float16)x[i];
  if (i < 512 * 1536)    wh[i] = (_Float16)w[i];
}

// dw[h][l] = <r_w_bias[h], pos[l]>
__global__ void dw_kernel(const float* __restrict__ rw, const _Float16* __restrict__ posh,
                          float* __restrict__ dw) {
  const int idx = blockIdx.x * 256 + threadIdx.x;
  if (idx >= 8 * 1024) return;
  const int h = idx >> 10, l = idx & 1023;
  float s = 0.f;
#pragma unroll 8
  for (int d = 0; d < 64; ++d) s += rw[h * 64 + d] * (float)posh[l * 64 + d];
  dw[idx] = s;
}

// ck[bh][j] = <r_r_bias[h], k[b,h,j,:]>
__global__ void ck_kernel(const float* __restrict__ rr, const _Float16* __restrict__ kh,
                          float* __restrict__ ck) {
  const int idx = blockIdx.x * 256 + threadIdx.x;
  if (idx >= 64 * 512) return;
  const int h = (idx >> 9) & 7;
  const _Float16* kp = kh + (long)idx * 64;
  float s = 0.f;
#pragma unroll 8
  for (int d = 0; d < 64; ++d) s += rr[h * 64 + d] * (float)kp[d];
  ck[idx] = s;
}

// qkv = xh[4096,512] @ wh[512,1536]; scatter f16 results to q/k/v in [b][h][l][d] layout.
__global__ __launch_bounds__(256) void qkv_gemm(const _Float16* __restrict__ xh,
                                                const _Float16* __restrict__ wh,
                                                _Float16* __restrict__ qh,
                                                _Float16* __restrict__ kh,
                                                _Float16* __restrict__ vh) {
  __shared__ __align__(16) _Float16 sA[128 * 32];
  __shared__ __align__(16) _Float16 sBT[128 * 32];   // B tile stored transposed [n][k]
  const int tid = threadIdx.x;
  const int wave = tid >> 5, lane = tid & 31;
  const int bn = blockIdx.x, bm = blockIdx.y;
  const int mw = (wave & 3) * 32, nw = (wave >> 2) * 64;
  const int nl = lane & 15, hi = (lane >> 4) << 3;

  v8f acc[2][4] = {};

  for (int kk = 0; kk < 512; kk += 32) {
    __syncthreads();
    for (int c = tid; c < 512; c += 256)             // A tile: 128x32 halfs (async -> LDS)
      cp16((uint4*)sA + c, (const uint4*)(xh + (bm * 128 + (c >> 2)) * 512 + kk) + (c & 3));
    for (int c = tid; c < 512; c += 256) {           // B tile: 32x128, transpose on store
      const int kr = c >> 4, q8 = c & 15;
      uint4 vv = *(const uint4*)(wh + (kk + kr) * 1536 + bn * 128 + q8 * 8);
      const _Float16* hv = (const _Float16*)&vv;
#pragma unroll
      for (int e = 0; e < 8; ++e) sBT[(q8 * 8 + e) * 32 + kr] = hv[e];
    }
    cp_fence();
    __syncthreads();
    v16h af[2], bf[4];
#pragma unroll
    for (int mt = 0; mt < 2; ++mt) af[mt] = load_frag(sA, mw + mt * 16 + nl, 32, lane);
#pragma unroll
    for (int nt = 0; nt < 4; ++nt) bf[nt] = load_frag(sBT, nw + nt * 16 + nl, 32, lane);
#pragma unroll
    for (int mt = 0; mt < 2; ++mt)
#pragma unroll
      for (int nt = 0; nt < 4; ++nt) acc[mt][nt] = wmma16(af[mt], bf[nt], acc[mt][nt]);
  }
#pragma unroll
  for (int mt = 0; mt < 2; ++mt)
#pragma unroll
    for (int nt = 0; nt < 4; ++nt)
#pragma unroll
      for (int v = 0; v < 8; ++v) {
        const int r = bm * 128 + mw + mt * 16 + v + hi;      // 0..4095
        const int cfull = bn * 128 + nw + nt * 16 + nl;      // 0..1535
        const int b = r >> 9, l = r & 511;
        const int which = cfull >> 9, c = cfull & 511;
        const int head = c >> 6, d = c & 63;
        _Float16* dst = (which == 0) ? qh : (which == 1) ? kh : vh;
        dst[(((long)((b << 3) + head) * 512 + l) << 6) + d] = (_Float16)acc[mt][nt][v];
      }
}

// Fused relative attention per (b,h, 64-row query tile).
// S[i,j] = <q_i,k_j> + ck[j] + dw[512+j-i] + <q_i,p_{j-i}> + <k_j,p_{i-j}>,
// then softmax over j, then O = P @ V.
__global__ __launch_bounds__(256) void attn_kernel(
    const _Float16* __restrict__ qh, const _Float16* __restrict__ kh,
    const _Float16* __restrict__ vh, const _Float16* __restrict__ posh,
    const float* __restrict__ dw, const float* __restrict__ ck,
    const int* __restrict__ mask, float* __restrict__ out) {
  __shared__ __align__(16) _Float16 sQ[64 * 64];     //  8 KB
  __shared__ __align__(16) _Float16 sKV[64 * 64];    //  8 KB  (K block row-major / V block transposed)
  __shared__ __align__(16) _Float16 sP[128 * 64];    // 16 KB  (127-row diagonal band of pos)
  __shared__ __align__(16) _Float16 sBt[64 * 128];   // 16 KB
  __shared__ __align__(16) _Float16 sEt[64 * 128];   // 16 KB
  __shared__ __align__(16) _Float16 sS[64 * 512];    // 64 KB  scores -> probabilities
  __shared__ float sred[256];
  __shared__ float rmax[64];
  __shared__ float srow[64];

  const int tid = threadIdx.x, wave = tid >> 5, lane = tid & 31;
  const int nl = lane & 15, hi = (lane >> 4) << 3;
  const int it = blockIdx.x, bh = blockIdx.y;
  const int b = bh >> 3, h = bh & 7;
  const int i0 = it * 64;
  const int m0 = (wave & 3) * 16;   // wave's 16-row strip
  const int nb = wave >> 2;         // wave's column half

  {
    const uint4* Qg = (const uint4*)(qh + ((long)bh * 512 + i0) * 64);
    for (int c = tid; c < 512; c += 256) cp16((uint4*)sQ + c, Qg + c);
  }

  for (int jb = 0; jb < 8; ++jb) {
    const int j0 = jb * 64;
    __syncthreads();
    {
      const uint4* Kg = (const uint4*)(kh + ((long)bh * 512 + j0) * 64);
      for (int c = tid; c < 512; c += 256) cp16((uint4*)sKV + c, Kg + c);
    }
    {
      const int pstart = 512 + j0 - i0 - 63;   // B-term band
      for (int c = tid; c < 1024; c += 256) {
        int r = pstart + (c >> 3);
        r = r < 0 ? 0 : (r > 1023 ? 1023 : r);
        cp16((uint4*)sP + c, (const uint4*)(posh + r * 64) + (c & 7));
      }
    }
    cp_fence();
    __syncthreads();
    {
      v16h a0 = load_frag(sQ,      m0 + nl, 64, lane);
      v16h a1 = load_frag(sQ + 32, m0 + nl, 64, lane);
#pragma unroll
      for (int nt = 0; nt < 2; ++nt) {               // S0 = Q.K^T -> sS (f16)
        const int n0 = nb * 32 + nt * 16;
        v8f acc = {};
        acc = wmma16(a0, load_frag(sKV,      n0 + nl, 64, lane), acc);
        acc = wmma16(a1, load_frag(sKV + 32, n0 + nl, 64, lane), acc);
#pragma unroll
        for (int v = 0; v < 8; ++v)
          sS[(m0 + v + hi) * 512 + j0 + n0 + nl] = (_Float16)acc[v];
      }
#pragma unroll
      for (int nt = 0; nt < 4; ++nt) {               // Bt = Q.Pband^T [64x128]
        const int c0 = nb * 64 + nt * 16;
        v8f acc = {};
        acc = wmma16(a0, load_frag(sP,      c0 + nl, 64, lane), acc);
        acc = wmma16(a1, load_frag(sP + 32, c0 + nl, 64, lane), acc);
#pragma unroll
        for (int v = 0; v < 8; ++v)
          sBt[(m0 + v + hi) * 128 + c0 + nl] = (_Float16)acc[v];
      }
    }
    __syncthreads();
    {
      const int pstart = 512 + i0 - j0 - 63;   // E-term band
      for (int c = tid; c < 1024; c += 256) {
        int r = pstart + (c >> 3);
        r = r < 0 ? 0 : (r > 1023 ? 1023 : r);
        cp16((uint4*)sP + c, (const uint4*)(posh + r * 64) + (c & 7));
      }
    }
    cp_fence();
    __syncthreads();
    {
      v16h a0 = load_frag(sKV,      m0 + nl, 64, lane);
      v16h a1 = load_frag(sKV + 32, m0 + nl, 64, lane);
#pragma unroll
      for (int nt = 0; nt < 4; ++nt) {               // Et = K.Pband2^T [64x128]
        const int c0 = nb * 64 + nt * 16;
        v8f acc = {};
        acc = wmma16(a0, load_frag(sP,      c0 + nl, 64, lane), acc);
        acc = wmma16(a1, load_frag(sP + 32, c0 + nl, 64, lane), acc);
#pragma unroll
        for (int v = 0; v < 8; ++v)
          sEt[(m0 + v + hi) * 128 + c0 + nl] = (_Float16)acc[v];
      }
    }
    __syncthreads();
    // assemble: diagonal-band gathers + rank-1 bias terms, scale, mask
    for (int e = tid; e < 4096; e += 256) {
      const int di = e >> 6, dj = e & 63;
      const int i = i0 + di, j = j0 + dj;
      float s = (float)sS[di * 512 + j]
              + ck[bh * 512 + j]
              + dw[h * 1024 + 512 + j - i]
              + (float)sBt[di * 128 + dj - di + 63]
              + (float)sEt[dj * 128 + di - dj + 63];
      s *= 0.125f;  // 1/sqrt(64)
      if (mask[b * 512 + j] == 0) s = -65504.f;
      sS[di * 512 + j] = (_Float16)s;
    }
  }
  __syncthreads();
  // softmax, 4 threads per row (all 8 waves active); unnormalized probs + row sums
  {
    const int r = tid >> 2, c0 = (tid & 3) * 128;
    float m = -3.0e38f;
    for (int j = 0; j < 128; ++j) {
      const float v = (float)sS[r * 512 + c0 + j];
      m = v > m ? v : m;
    }
    sred[tid] = m;
  }
  __syncthreads();
  if ((tid & 3) == 0) {
    const float a = sred[tid]     > sred[tid + 1] ? sred[tid]     : sred[tid + 1];
    const float c = sred[tid + 2] > sred[tid + 3] ? sred[tid + 2] : sred[tid + 3];
    rmax[tid >> 2] = a > c ? a : c;
  }
  __syncthreads();
  {
    const int r = tid >> 2, c0 = (tid & 3) * 128;
    const float m = rmax[r];
    float sum = 0.f;
    for (int j = 0; j < 128; ++j) {
      const float p = __expf((float)sS[r * 512 + c0 + j] - m);
      sum += p;
      sS[r * 512 + c0 + j] = (_Float16)p;
    }
    sred[tid] = sum;
  }
  __syncthreads();
  if ((tid & 3) == 0)
    srow[tid >> 2] = sred[tid] + sred[tid + 1] + sred[tid + 2] + sred[tid + 3];
  // O = P @ V
  const int n0w = nb * 32;
  v8f o[2] = {};
  for (int jb = 0; jb < 8; ++jb) {
    __syncthreads();
    if (jb < 7)  // prefetch next V block (global_prefetch_b8)
      __builtin_prefetch((const char*)(vh + ((long)bh * 512 + (jb + 1) * 64) * 64) + tid * 32, 0, 3);
    {
      const uint4* Vg = (const uint4*)(vh + ((long)bh * 512 + jb * 64) * 64);
      for (int c = tid; c < 512; c += 256) {         // V block transposed -> sKV[d][j]
        const int r = c >> 3, q8 = c & 7;
        uint4 vv = Vg[c];
        const _Float16* hv = (const _Float16*)&vv;
#pragma unroll
        for (int e = 0; e < 8; ++e) sKV[(q8 * 8 + e) * 64 + r] = hv[e];
      }
    }
    __syncthreads();
#pragma unroll
    for (int ks = 0; ks < 2; ++ks) {
      v16h a = load_frag(sS + jb * 64 + ks * 32, m0 + nl, 512, lane);
#pragma unroll
      for (int nt = 0; nt < 2; ++nt)
        o[nt] = wmma16(a, load_frag(sKV + ks * 32, n0w + nt * 16 + nl, 64, lane), o[nt]);
    }
  }
  __syncthreads();
#pragma unroll
  for (int nt = 0; nt < 2; ++nt)
#pragma unroll
    for (int v = 0; v < 8; ++v) {
      const int m = m0 + v + hi;
      const int d = n0w + nt * 16 + nl;
      out[((long)(b * 512 + i0 + m)) * 512 + h * 64 + d] = o[nt][v] / srow[m];
    }
}

extern "C" void kernel_launch(void* const* d_in, const int* in_sizes, int n_in,
                              void* d_out, int out_size, void* d_ws, size_t ws_size,
                              hipStream_t stream) {
  (void)in_sizes; (void)n_in; (void)out_size; (void)ws_size;
  const float* x    = (const float*)d_in[0];
  const int*   mask = (const int*)d_in[1];
  const float* Wq   = (const float*)d_in[2];
  const float* rr   = (const float*)d_in[3];  // r_r_bias
  const float* rw   = (const float*)d_in[4];  // r_w_bias
  float* out = (float*)d_out;

  char* ws = (char*)d_ws;
  _Float16* posh = (_Float16*)(ws);                 // 1024*64*2        = 131072
  float*    dwv  = (float*)(ws + 131072);           // 8*1024*4         = 32768
  _Float16* xh   = (_Float16*)(ws + 163840);        // 2097152*2        = 4194304
  _Float16* wh   = (_Float16*)(ws + 4358144);       // 786432*2         = 1572864
  _Float16* qh   = (_Float16*)(ws + 5931008);       // 2097152*2
  _Float16* kh   = (_Float16*)(ws + 10125312);      // 2097152*2
  _Float16* vh   = (_Float16*)(ws + 14319616);      // 2097152*2
  float*    ckv  = (float*)(ws + 18513920);         // 64*512*4 -> total 18644992 B

  pos_kernel<<<1024, 64, 0, stream>>>(posh);
  cvt_kernel<<<8192, 256, 0, stream>>>(x, Wq, xh, wh);
  dw_kernel<<<32, 256, 0, stream>>>(rw, posh, dwv);
  qkv_gemm<<<dim3(12, 32), 256, 0, stream>>>(xh, wh, qh, kh, vh);
  ck_kernel<<<128, 256, 0, stream>>>(rr, kh, ckv);
  attn_kernel<<<dim3(8, 64), 256, 0, stream>>>(qh, kh, vh, posh, dwv, ckv, mask, out);
}
